// AttentionDTI_53042846105895
// MI455X (gfx1250) — matrix operand order repaired and checked by reading, not compiled, not run on hardware
//
#include <hip/hip_runtime.h>
#include <hip/hip_bf16.h>
#include <math.h>

// ---------------------------------------------------------------------------
// AttentionDTI forward, fused for MI455X (gfx1250, wave32, WMMA).
//
// (1) Exact algebraic collapse of the attention core: att = relu(d⊕p)@Wa.T+ba
//     is only consumed via means over i / over j; linearity moves the sums
//     inside Wa -> the 34-GFLOP GEMM + 426 MB tensors become two relu-sum
//     reductions (s, t) + two tiny 160x160 WMMA projections.
// (2) Convs as implicit-GEMM WMMA with zero-cost im2col: in length-major
//     layout inT[b][l][ci], the im2col row of output l with K-order (k,ci) is
//     inT_flat[l*Cin + kk] -> A = inT with row stride Cin (overlapping rows),
//     contiguous along K. Weights repacked once per call to f16
//     (co, k*Cin+ci), zero-padded in K (mult of 32) and N (mult of 16).
// (3) A-reuse: each wave owns a 16 x (NT*16) output strip with NT register
//     accumulators; A is loaded/converted once per K-chunk and feeds NT
//     independent back-to-back WMMAs (good XDL pipelining, ~5x less L2 A
//     traffic on the big GEMMs).
// All matmuls share one templated V_WMMA_F32_16X16X32_F16 kernel
// (f16 A/B, f32 acc), b128 loads, branchless OOB, full EXEC at WMMA.
// ---------------------------------------------------------------------------

typedef __attribute__((ext_vector_type(16))) _Float16 v16h;
typedef __attribute__((ext_vector_type(8)))  _Float16 h8;
typedef __attribute__((ext_vector_type(8)))  float    v8f;

#define B_   8
#define DIM_ 64
#define C_   160
#define LI_  85    // drug conv3 output length
#define LJ_  979   // protein conv3 output length
#define PAD_ 64    // per-buffer slack (floats) for overlapping/padded-K reads

// -------- embedding gather (length-major): out[b][l][c] = emb[idx[b][l]][c] -
__global__ void embed_kernel(const int* __restrict__ idx,
                             const float* __restrict__ emb,
                             float* __restrict__ out, int L) {
  int i = blockIdx.x * blockDim.x + threadIdx.x;
  int total = B_ * L * DIM_;
  if (i >= total) return;
  int c = i & (DIM_ - 1);
  int l = (i >> 6) % L;
  int b = i / (DIM_ * L);
  out[i] = emb[idx[b * L + l] * DIM_ + c];
}

// -------- weight packing to f16 -------------------------------------------
__global__ void pack_linear_kernel(const float* __restrict__ W,
                                   _Float16* __restrict__ W16,
                                   int N, int NP, int K) {
  int i = blockIdx.x * blockDim.x + threadIdx.x;
  if (i >= NP * K) return;
  int n = i / K;
  W16[i] = (n < N) ? (_Float16)W[i] : (_Float16)0.0f;
}
// conv: W16[n][k*Cin+ci] = w[n][ci][k]; zero for n>=N or k>=Kw (K padding).
__global__ void pack_conv_kernel(const float* __restrict__ w,
                                 _Float16* __restrict__ W16,
                                 int N, int NP, int Cin, int Kw, int KT) {
  int i = blockIdx.x * blockDim.x + threadIdx.x;
  if (i >= NP * KT) return;
  int n  = i / KT;
  int kk = i % KT;
  int k  = kk / Cin;
  int ci = kk % Cin;
  float v = (n < N && k < Kw) ? w[((size_t)n * Cin + ci) * Kw + k] : 0.0f;
  W16[i] = (_Float16)v;
}

// -------- unified WMMA GEMM -----------------------------------------------
// Y[b][m][n] = epi( sum_kk X[b][m*xrstride + kk]*xs * W16[n][kk] + bias[n] )
// KT = KCH*32. A rows may overlap (convs: xrstride=Cin). W16 padded [NP][KT].
// 4 waves/block; each wave computes a 16 x (NT*16) strip (NT accumulators),
// loading/converting A once per K-chunk. NP/16 must be a multiple of NT.
// A-layout (ISA 7.12.2, 16-bit A 16x32): lanes 0-15 row M=lane, elems 0..7 ->
// K=kc..kc+7, elems 8..15 -> K=kc+16..kc+23; lanes 16-31 same rows, +8 on K.
// B-layout (16-bit B): lanes 0-15 col N=lane, elems 0..7 -> K=kc..kc+7,
// elems 8..15 -> K=kc+16..kc+23; lanes 16-31 same cols, +8 on K.
// epi: 0 = identity, 1 = sigmoid, 2 = relu.
template <int KCH, int NT>
__global__ void wmma_gemm_kernel(const float* __restrict__ X, size_t xbstride,
                                 int xrstride,
                                 const _Float16* __restrict__ W16,
                                 const float* __restrict__ bias,
                                 float* __restrict__ Y, size_t ybstride,
                                 int M, int N, int NP, float x_scale, int epi) {
  constexpr int KT = KCH * 32;
  const int ngroups = (NP >> 4) / NT;
  const int b       = blockIdx.y;
  const int wave    = threadIdx.x >> 5;
  const int tile    = blockIdx.x * 4 + wave;
  const int tilesM  = (M + 15) >> 4;
  if (tile >= tilesM * ngroups) return;     // wave-uniform exit
  const int m0     = (tile / ngroups) * 16;
  const int n0base = (tile % ngroups) * (NT * 16);
  const int lane   = threadIdx.x & 31;
  const int half   = lane >> 4;
  const int r      = lane & 15;

  const float* Xb = X + (size_t)b * xbstride;
  float*       Yb = Y + (size_t)b * ybstride;

  const int   arow   = m0 + r;
  const bool  avalid = (arow < M);
  const float lscale = avalid ? x_scale : 0.0f;   // branchless OOB masking
  const float* xrow  = Xb + (size_t)(avalid ? arow : 0) * xrstride;
  const _Float16* wbase = W16 + (size_t)(n0base + r) * KT + half * 8;

  v8f acc[NT];
#pragma unroll
  for (int nt = 0; nt < NT; ++nt) acc[nt] = (v8f){};

  auto step = [&](int kc) {
    const int ka = kc + half * 8;    // A: this lane's first K run
    const float4 a0 = *(const float4*)(xrow + ka);
    const float4 a1 = *(const float4*)(xrow + ka + 4);
    const float4 a2 = *(const float4*)(xrow + ka + 16);
    const float4 a3 = *(const float4*)(xrow + ka + 20);
    v16h A;
    A[0]  = (_Float16)(a0.x * lscale); A[1]  = (_Float16)(a0.y * lscale);
    A[2]  = (_Float16)(a0.z * lscale); A[3]  = (_Float16)(a0.w * lscale);
    A[4]  = (_Float16)(a1.x * lscale); A[5]  = (_Float16)(a1.y * lscale);
    A[6]  = (_Float16)(a1.z * lscale); A[7]  = (_Float16)(a1.w * lscale);
    A[8]  = (_Float16)(a2.x * lscale); A[9]  = (_Float16)(a2.y * lscale);
    A[10] = (_Float16)(a2.z * lscale); A[11] = (_Float16)(a2.w * lscale);
    A[12] = (_Float16)(a3.x * lscale); A[13] = (_Float16)(a3.y * lscale);
    A[14] = (_Float16)(a3.z * lscale); A[15] = (_Float16)(a3.w * lscale);
    const _Float16* wp = wbase + kc;
#pragma unroll
    for (int nt = 0; nt < NT; ++nt) {
      const h8 blo = *(const h8*)(wp);
      const h8 bhi = *(const h8*)(wp + 16);
      v16h Bv = __builtin_shufflevector(blo, bhi, 0, 1, 2, 3, 4, 5, 6, 7,
                                        8, 9, 10, 11, 12, 13, 14, 15);
      acc[nt] = __builtin_amdgcn_wmma_f32_16x16x32_f16(
          false, A, false, Bv, (short)0, acc[nt], false, false);
      wp += (size_t)16 * KT;
    }
  };
  if constexpr (KCH <= 8) {
#pragma unroll
    for (int c = 0; c < KCH; ++c) step(c * 32);
  } else {
#pragma unroll 2
    for (int c = 0; c < KCH; ++c) step(c * 32);
  }

#pragma unroll
  for (int nt = 0; nt < NT; ++nt) {
    const int   n  = n0base + nt * 16 + r;
    const float bv = (n < N) ? bias[n] : 0.0f;
#pragma unroll
    for (int v = 0; v < 8; ++v) {
      int m = m0 + v + half * 8;  // D: lanes 0-15 -> M=v, 16-31 -> M=v+8
      if (m < M && n < N) {
        float y = acc[nt][v] + bv;
        if (epi == 1)      y = 1.0f / (1.0f + __expf(-y));
        else if (epi == 2) y = fmaxf(y, 0.0f);
        Yb[(size_t)m * N + n] = y;
      }
    }
  }
}

// -------- pairwise relu-sum reduction (no atomics):
// out[b][o][c] = sum_n relu(Aout[b][o][c] + Ain[b][n][c]);  block=(o,b), 160 thr
__global__ void pairsum_kernel(const float* __restrict__ Aout,
                               const float* __restrict__ Ain,
                               float* __restrict__ out, int LO, int LN) {
  int c = threadIdx.x;
  int o = blockIdx.x;
  int b = blockIdx.y;
  float d = Aout[((size_t)b * LO + o) * C_ + c];
  const float* p = Ain + (size_t)b * LN * C_ + c;
  float acc = 0.0f;
  for (int n = 0; n < LN; ++n) acc += fmaxf(d + p[(size_t)n * C_], 0.0f);
  out[((size_t)b * LO + o) * C_ + c] = acc;
}

// -------- gate, scale, global max-pool, concat (length-major inputs) -------
__global__ void scale_pool_kernel(const float* __restrict__ dcT,
                                  const float* __restrict__ c_att,
                                  const float* __restrict__ pcT,
                                  const float* __restrict__ pr_att,
                                  float* __restrict__ xcat) {
  int b = blockIdx.x;
  int o = threadIdx.x;  // 320 threads
  float m = -INFINITY;
  if (o < C_) {
    for (int i = 0; i < LI_; ++i) {
      size_t ix = ((size_t)b * LI_ + i) * C_ + o;
      m = fmaxf(m, dcT[ix] * (0.5f + c_att[ix]));
    }
  } else {
    int oo = o - C_;
    for (int j = 0; j < LJ_; ++j) {
      size_t ix = ((size_t)b * LJ_ + j) * C_ + oo;
      m = fmaxf(m, pcT[ix] * (0.5f + pr_att[ix]));
    }
  }
  xcat[b * 320 + o] = m;
}

// -------- final MLP 320->1024->1024->512->1, leaky_relu(0.01) --------------
__global__ void mlp_kernel(const float* __restrict__ xcat,
                           const float* __restrict__ fW1, const float* __restrict__ fb1,
                           const float* __restrict__ fW2, const float* __restrict__ fb2,
                           const float* __restrict__ fW3, const float* __restrict__ fb3,
                           const float* __restrict__ fW4, const float* __restrict__ fb4,
                           float* __restrict__ out) {
  __shared__ float bufA[1024];
  __shared__ float bufB[1024];
  __shared__ float red[256];
  int b = blockIdx.x, t = threadIdx.x;
  if (t < 320) bufA[t] = xcat[b * 320 + t];
  __syncthreads();
  float a = fb1[t];
  for (int k = 0; k < 320; ++k) a = fmaf(fW1[(size_t)t * 320 + k], bufA[k], a);
  bufB[t] = a > 0.0f ? a : 0.01f * a;
  __syncthreads();
  a = fb2[t];
  for (int k = 0; k < 1024; ++k) a = fmaf(fW2[(size_t)t * 1024 + k], bufB[k], a);
  bufA[t] = a > 0.0f ? a : 0.01f * a;
  __syncthreads();
  if (t < 512) {
    a = fb3[t];
    for (int k = 0; k < 1024; ++k) a = fmaf(fW3[(size_t)t * 1024 + k], bufA[k], a);
    bufB[t] = a > 0.0f ? a : 0.01f * a;
  }
  __syncthreads();
  if (t < 256) red[t] = fW4[t] * bufB[t] + fW4[t + 256] * bufB[t + 256];
  __syncthreads();
  for (int st = 128; st > 0; st >>= 1) {
    if (t < st) red[t] += red[t + st];
    __syncthreads();
  }
  if (t == 0) out[b] = red[0] + fb4[0];
}

// ---------------------------------------------------------------------------
extern "C" void kernel_launch(void* const* d_in, const int* in_sizes, int n_in,
                              void* d_out, int out_size, void* d_ws, size_t ws_size,
                              hipStream_t stream) {
  (void)in_sizes; (void)n_in; (void)out_size; (void)ws_size;
  const int*   drug     = (const int*)  d_in[0];
  const int*   protein  = (const int*)  d_in[1];
  const float* drug_emb = (const float*)d_in[2];
  const float* prot_emb = (const float*)d_in[3];
  const float* dW1 = (const float*)d_in[4];  const float* db1 = (const float*)d_in[5];
  const float* dW2 = (const float*)d_in[6];  const float* db2 = (const float*)d_in[7];
  const float* dW3 = (const float*)d_in[8];  const float* db3 = (const float*)d_in[9];
  const float* pW1 = (const float*)d_in[10]; const float* pb1 = (const float*)d_in[11];
  const float* pW2 = (const float*)d_in[12]; const float* pb2 = (const float*)d_in[13];
  const float* pW3 = (const float*)d_in[14]; const float* pb3 = (const float*)d_in[15];
  const float* Wda = (const float*)d_in[16]; const float* bda = (const float*)d_in[17];
  const float* Wpa = (const float*)d_in[18]; const float* bpa = (const float*)d_in[19];
  const float* Wa  = (const float*)d_in[20]; const float* ba  = (const float*)d_in[21];
  const float* fW1 = (const float*)d_in[22]; const float* fb1 = (const float*)d_in[23];
  const float* fW2 = (const float*)d_in[24]; const float* fb2 = (const float*)d_in[25];
  const float* fW3 = (const float*)d_in[26]; const float* fb3 = (const float*)d_in[27];
  const float* fW4 = (const float*)d_in[28]; const float* fb4 = (const float*)d_in[29];
  float* out = (float*)d_out;

  // ---- workspace layout (floats; all offsets multiples of 4 -> 16B aligned)
  float* w = (float*)d_ws;
  size_t off = 0;
  float* deT    = w + off; off += (size_t)B_ * 100  * DIM_ + PAD_;
  float* peT    = w + off; off += (size_t)B_ * 1000 * DIM_ + PAD_;
  float* d1T    = w + off; off += (size_t)B_ * 97   * 40   + PAD_;
  float* d2T    = w + off; off += (size_t)B_ * 92   * 80   + PAD_;
  float* dcT    = w + off; off += (size_t)B_ * LI_  * C_   + PAD_;
  float* p1T    = w + off; off += (size_t)B_ * 997  * 40   + PAD_;
  float* p2T    = w + off; off += (size_t)B_ * 990  * 80   + PAD_;
  float* pcT    = w + off; off += (size_t)B_ * LJ_  * C_   + PAD_;
  float* d_att  = w + off; off += (size_t)B_ * LI_ * C_;
  float* p_att  = w + off; off += (size_t)B_ * LJ_ * C_;
  float* s_buf  = w + off; off += (size_t)B_ * LI_ * C_;
  float* t_buf  = w + off; off += (size_t)B_ * LJ_ * C_;
  float* c_att  = w + off; off += (size_t)B_ * LI_ * C_;
  float* pr_att = w + off; off += (size_t)B_ * LJ_ * C_;
  float* xcat   = w + off; off += (size_t)B_ * 320;
  // f16 packed weights (counts in halves, all multiples of 8 -> 16B aligned)
  _Float16* h = (_Float16*)(w + off);
  size_t ho = 0;
  _Float16* wp1d  = h + ho; ho += (size_t)48  * 256;
  _Float16* wp2d  = h + ho; ho += (size_t)80  * 256;
  _Float16* wp3d  = h + ho; ho += (size_t)160 * 640;
  _Float16* wp1p  = h + ho; ho += (size_t)48  * 256;
  _Float16* wp2p  = h + ho; ho += (size_t)80  * 320;
  _Float16* wp3p  = h + ho; ho += (size_t)160 * 960;
  _Float16* wda16 = h + ho; ho += (size_t)160 * 160;
  _Float16* wpa16 = h + ho; ho += (size_t)160 * 160;
  _Float16* wa16  = h + ho; ho += (size_t)160 * 160;

  const int TB = 256;
  #define GRID1(n) dim3(((n) + TB - 1) / TB)
  // grid for a 16x(NT*16)-strip wave decomposition
  #define GGEMM(M, NP, NT) dim3((((((M) + 15) / 16) * (((NP) / 16) / (NT))) + 3) / 4, B_)

  // embeddings (length-major)
  embed_kernel<<<GRID1(B_ * 100  * DIM_), TB, 0, stream>>>(drug,    drug_emb, deT, 100);
  embed_kernel<<<GRID1(B_ * 1000 * DIM_), TB, 0, stream>>>(protein, prot_emb, peT, 1000);

  // weight packing (f16, padded)
  pack_conv_kernel<<<GRID1(48  * 256), TB, 0, stream>>>(dW1, wp1d, 40, 48, 64, 4, 256);
  pack_conv_kernel<<<GRID1(80  * 256), TB, 0, stream>>>(dW2, wp2d, 80, 80, 40, 6, 256);
  pack_conv_kernel<<<GRID1(160 * 640), TB, 0, stream>>>(dW3, wp3d, 160, 160, 80, 8, 640);
  pack_conv_kernel<<<GRID1(48  * 256), TB, 0, stream>>>(pW1, wp1p, 40, 48, 64, 4, 256);
  pack_conv_kernel<<<GRID1(80  * 320), TB, 0, stream>>>(pW2, wp2p, 80, 80, 40, 8, 320);
  pack_conv_kernel<<<GRID1(160 * 960), TB, 0, stream>>>(pW3, wp3p, 160, 160, 80, 12, 960);
  pack_linear_kernel<<<GRID1(160 * 160), TB, 0, stream>>>(Wda, wda16, 160, 160, 160);
  pack_linear_kernel<<<GRID1(160 * 160), TB, 0, stream>>>(Wpa, wpa16, 160, 160, 160);
  pack_linear_kernel<<<GRID1(160 * 160), TB, 0, stream>>>(Wa,  wa16,  160, 160, 160);

  // conv stacks as implicit-GEMM WMMA (epi=2: relu)
  wmma_gemm_kernel<8, 3> <<<GGEMM(97, 48, 3),   128, 0, stream>>>(deT, 100 * 64, 64, wp1d, db1, d1T, (size_t)97 * 40,  97,  40, 48,  1.0f, 2);
  wmma_gemm_kernel<8, 5> <<<GGEMM(92, 80, 5),   128, 0, stream>>>(d1T, (size_t)97 * 40, 40, wp2d, db2, d2T, (size_t)92 * 80,  92,  80, 80,  1.0f, 2);
  wmma_gemm_kernel<20, 5><<<GGEMM(LI_, 160, 5), 128, 0, stream>>>(d2T, (size_t)92 * 80, 80, wp3d, db3, dcT, (size_t)LI_ * C_, LI_, 160, 160, 1.0f, 2);
  wmma_gemm_kernel<8, 3> <<<GGEMM(997, 48, 3),  128, 0, stream>>>(peT, 1000 * 64, 64, wp1p, pb1, p1T, (size_t)997 * 40, 997, 40, 48,  1.0f, 2);
  wmma_gemm_kernel<10, 5><<<GGEMM(990, 80, 5),  128, 0, stream>>>(p1T, (size_t)997 * 40, 40, wp2p, pb2, p2T, (size_t)990 * 80, 990, 80, 80,  1.0f, 2);
  wmma_gemm_kernel<30, 5><<<GGEMM(LJ_, 160, 5), 128, 0, stream>>>(p2T, (size_t)990 * 80, 80, wp3p, pb3, pcT, (size_t)LJ_ * C_, LJ_, 160, 160, 1.0f, 2);

  // attention projections (epi=0)
  wmma_gemm_kernel<5, 5><<<GGEMM(LI_, 160, 5), 128, 0, stream>>>(dcT, (size_t)LI_ * C_, C_, wda16, bda, d_att, (size_t)LI_ * C_, LI_, 160, 160, 1.0f, 0);
  wmma_gemm_kernel<5, 5><<<GGEMM(LJ_, 160, 5), 128, 0, stream>>>(pcT, (size_t)LJ_ * C_, C_, wpa16, bpa, p_att, (size_t)LJ_ * C_, LJ_, 160, 160, 1.0f, 0);

  // pairwise relu sums: s[i,c] = sum_j relu(d+p) ; t[j,c] = sum_i relu(d+p)
  pairsum_kernel<<<dim3(LI_, B_), C_, 0, stream>>>(d_att, p_att, s_buf, LI_, LJ_);
  pairsum_kernel<<<dim3(LJ_, B_), C_, 0, stream>>>(p_att, d_att, t_buf, LJ_, LI_);

  // gates (epi=1: sigmoid; mean folded into f16 conversion scale)
  wmma_gemm_kernel<5, 5><<<GGEMM(LI_, 160, 5), 128, 0, stream>>>(s_buf, (size_t)LI_ * C_, C_, wa16, ba, c_att,  (size_t)LI_ * C_, LI_, 160, 160, 1.0f / (float)LJ_, 1);
  wmma_gemm_kernel<5, 5><<<GGEMM(LJ_, 160, 5), 128, 0, stream>>>(t_buf, (size_t)LJ_ * C_, C_, wa16, ba, pr_att, (size_t)LJ_ * C_, LJ_, 160, 160, 1.0f / (float)LI_, 1);

  // gate-scale + global max-pool + concat (length-major)
  scale_pool_kernel<<<dim3(B_), 320, 0, stream>>>(dcT, c_att, pcT, pr_att, xcat);

  // final MLP
  mlp_kernel<<<dim3(B_), 1024, 0, stream>>>(xcat, fW1, fb1, fW2, fb2, fW3, fb3, fW4, fb4, out);
  #undef GRID1
  #undef GGEMM
}
